// Decoder_15401752723792
// MI455X (gfx1250) — compile-verified, matrix operand order checked by reading, not run on hardware
//
#include <hip/hip_runtime.h>

// ---------------- problem constants ----------------
constexpr int kD   = 1024;   // model dim
constexpr int kH   = 16;     // heads
constexpr int kDH  = 64;     // head dim
constexpr int kFF  = 4096;   // ffn dim
constexpr int kL   = 6;      // layers
constexpr int kB   = 8;      // batch
constexpr int kN   = 512;    // sequence length (q and k)
constexpr int kM   = kB * kN; // 4096 rows

typedef __attribute__((ext_vector_type(16))) __bf16    v16bf;
typedef __attribute__((ext_vector_type(2)))  __bf16    v2bf;
typedef __attribute__((ext_vector_type(8)))  float     v8f;
typedef __attribute__((ext_vector_type(4)))  unsigned  v4u;

union BFrag { v16bf v; unsigned u[8]; v4u q[2]; };

#if defined(__has_builtin)
#if __has_builtin(__builtin_amdgcn_cvt_pk_bf16_f32)
#define HAVE_CVT_PK_BF16 1
#endif
#endif

__device__ __forceinline__ unsigned short f2bf(float f) {
    union { float f; unsigned u; } x; x.f = f;
    unsigned r = x.u + 0x7FFFu + ((x.u >> 16) & 1u);  // round-to-nearest-even
    return (unsigned short)(r >> 16);
}

__device__ __forceinline__ unsigned pack2bf(float f0, float f1) {
#ifdef HAVE_CVT_PK_BF16
    union { v2bf v; unsigned u; } x;
    x.v = __builtin_amdgcn_cvt_pk_bf16_f32(f0, f1);
    return x.u;
#else
    return (unsigned)f2bf(f0) | ((unsigned)f2bf(f1) << 16);
#endif
}

__device__ __forceinline__ v8f vzero8() {
    v8f z = {0.f,0.f,0.f,0.f,0.f,0.f,0.f,0.f};
    return z;
}

// ============================================================
// GEMM: C[M,N] = A[M,K] @ B[K,N] + bias[N]  (optional ReLU)
// fp32 in/out, bf16 WMMA compute. Block = 128 threads (4 waves),
// tile 64(M) x 64(N), BK = 32. Software-pipelined global loads;
// LDS row stride 40 shorts (80B) so fragments load as 2x b128.
// ============================================================
__global__ __launch_bounds__(128)
void gemm_bf16(const float* __restrict__ A, const float* __restrict__ Bw,
               const float* __restrict__ bias, float* __restrict__ C,
               int M, int K, int N, int relu)
{
    __shared__ __align__(16) unsigned short As[64 * 40];   // [m][k]
    __shared__ __align__(16) unsigned short Bs[64 * 40];   // [n][k] (B transposed)

    const int tid  = threadIdx.x;
    const int w    = tid >> 5;      // wave 0..3 -> 16 rows each
    const int l    = tid & 31;
    const int hi   = (l >> 4) & 1;  // lane half (K offset per ISA layout)
    const int ln16 = l & 15;
    const int m0   = blockIdx.x * 64;
    const int n0   = blockIdx.y * 64;

    v8f acc[4];
    #pragma unroll
    for (int j = 0; j < 4; ++j) acc[j] = vzero8();

    unsigned*  As32 = (unsigned*)As;
    unsigned*  Bs32 = (unsigned*)Bs;
    const v4u* AsQ  = (const v4u*)As;    // row stride 5 (80B)
    const v4u* BsQ  = (const v4u*)Bs;

    unsigned aP[8], bP[8];

    // prologue: load first tile into registers (bf16-packed pairs)
    {
        #pragma unroll
        for (int i = 0; i < 8; ++i) {
            int idx = i * 128 + tid;           // 1024 k-pairs of A (64 rows x 16)
            int r = idx >> 4, kp = idx & 15;
            const float* p = &A[(size_t)(m0 + r) * K + 2 * kp];
            aP[i] = pack2bf(p[0], p[1]);
        }
        #pragma unroll
        for (int i = 0; i < 8; ++i) {
            int idx = i * 128 + tid;           // 1024 pairs of B (64 n x 16 kp)
            int n = idx & 63, kp = idx >> 6;
            const float* p = &Bw[(size_t)(2 * kp) * N + n0 + n];
            bP[i] = pack2bf(p[0], p[N]);
        }
    }

    for (int kt = 0; kt < K; kt += 32) {
        __syncthreads();   // previous compute done reading LDS
        #pragma unroll
        for (int i = 0; i < 8; ++i) {
            int idx = i * 128 + tid;
            int r = idx >> 4, kp = idx & 15;
            As32[r * 20 + kp] = aP[i];
        }
        #pragma unroll
        for (int i = 0; i < 8; ++i) {
            int idx = i * 128 + tid;
            int n = idx & 63, kp = idx >> 6;
            Bs32[n * 20 + kp] = bP[i];
        }
        __syncthreads();

        // prefetch next tile into registers while WMMAs run
        if (kt + 32 < K) {
            #pragma unroll
            for (int i = 0; i < 8; ++i) {
                int idx = i * 128 + tid;
                int r = idx >> 4, kp = idx & 15;
                const float* p = &A[(size_t)(m0 + r) * K + kt + 32 + 2 * kp];
                aP[i] = pack2bf(p[0], p[1]);
            }
            #pragma unroll
            for (int i = 0; i < 8; ++i) {
                int idx = i * 128 + tid;
                int n = idx & 63, kp = idx >> 6;
                const float* p = &Bw[(size_t)(kt + 32 + 2 * kp) * N + n0 + n];
                bP[i] = pack2bf(p[0], p[N]);
            }
        }

        // fragments: 2x ds_load_b128 each, all loaded before the WMMA chain
        BFrag af, bfr[4];
        {
            int ar = (w * 16 + ln16) * 5;
            af.q[0] = AsQ[ar + hi];
            af.q[1] = AsQ[ar + 2 + hi];
        }
        #pragma unroll
        for (int j = 0; j < 4; ++j) {
            int br = (j * 16 + ln16) * 5;
            bfr[j].q[0] = BsQ[br + hi];
            bfr[j].q[1] = BsQ[br + 2 + hi];
        }
        #pragma unroll
        for (int j = 0; j < 4; ++j) {
            acc[j] = __builtin_amdgcn_wmma_f32_16x16x32_bf16(
                false, af.v, false, bfr[j].v, (short)0, acc[j], false, false);
        }
    }

    #pragma unroll
    for (int j = 0; j < 4; ++j) {
        int n = n0 + j * 16 + ln16;
        float bv = bias ? bias[n] : 0.f;
        #pragma unroll
        for (int i = 0; i < 8; ++i) {
            int m = m0 + w * 16 + i + (hi << 3);
            float val = acc[j][i] + bv;
            if (relu) val = fmaxf(val, 0.f);
            C[(size_t)m * N + n] = val;
        }
    }
}

// ============================================================
// Attention: per (b, h, 32-query tile).
//   S = Q K^T / 8  (mask==0 -> -10000), softmax, O = P V
// Q/K/V are [b*n, D] with head columns h*64..h*64+63.
// ============================================================
constexpr int kQT  = 32;
constexpr int kSMEM_S = 65536;              // 32*512 f32
constexpr int kSMEM_P = 32768;              // 32*512 bf16 (row = 1024B, 16B aligned)
constexpr int kSMEM_Q = 32 * 72 * 2;        // 4608 (row 144B, 16B aligned)
constexpr int kSMEM_V = 64 * 136 * 2;       // 17408 (row 272B, 16B aligned)
constexpr int kSMEM_R = 32 * 4 * 4;         // 512
constexpr size_t kSMEM_ATTN = kSMEM_S + kSMEM_P + kSMEM_Q + kSMEM_V + kSMEM_R;

__global__ __launch_bounds__(128)
void attn_kernel(const float* __restrict__ Qb, const float* __restrict__ Kb,
                 const float* __restrict__ Vb, const int* __restrict__ mask,
                 float* __restrict__ AO)
{
    extern __shared__ __align__(16) char smem[];
    float*          Sbuf = (float*)smem;
    unsigned short* Pbuf = (unsigned short*)(smem + kSMEM_S);
    unsigned short* Qs   = (unsigned short*)(smem + kSMEM_S + kSMEM_P);
    unsigned short* Vt   = (unsigned short*)(smem + kSMEM_S + kSMEM_P + kSMEM_Q);
    float*          red  = (float*)(smem + kSMEM_S + kSMEM_P + kSMEM_Q + kSMEM_V);

    const int tid  = threadIdx.x;
    const int w    = tid >> 5;
    const int l    = tid & 31;
    const int hi   = (l >> 4) & 1;
    const int ln16 = l & 15;

    const int bx    = blockIdx.x;
    const int qt    = bx & 15;            // 512/32 query tiles
    const int hh    = (bx >> 4) & 15;     // head
    const int b     = bx >> 8;            // batch
    const int qbase = qt * kQT;
    const int rowb  = b * kN;
    const int h64   = hh * kDH;

    // stage Q tile 32x64 as bf16 (packed-pair stores)
    #pragma unroll
    for (int i = 0; i < 8; ++i) {
        int idx = i * 128 + tid;               // 1024 pairs
        int r = idx >> 5, cp = idx & 31;       // 32 pairs per row
        const float* p = &Qb[(size_t)(rowb + qbase + r) * kD + h64 + 2 * cp];
        ((unsigned*)Qs)[r * 36 + cp] = pack2bf(p[0], p[1]);
    }
    __syncthreads();

    const v4u* QsQ = (const v4u*)Qs;           // row stride 9 (144B)

    // ---- phase 1: scores (each wave owns 128 keys) ----
    for (int kk = 0; kk < 8; ++kk) {
        int knb = w * 128 + kk * 16;
        v8f s0 = vzero8(), s1 = vzero8();
        #pragma unroll
        for (int ks = 0; ks < 2; ++ks) {
            BFrag a0, a1, bfr;
            {
                int base = ks * 4 + hi;
                a0.q[0] = QsQ[(ln16) * 9 + base];
                a0.q[1] = QsQ[(ln16) * 9 + base + 2];
                a1.q[0] = QsQ[(16 + ln16) * 9 + base];
                a1.q[1] = QsQ[(16 + ln16) * 9 + base + 2];
            }
            int key = knb + ln16;
            #pragma unroll
            for (int v = 0; v < 8; ++v) {
                int kd = ((v >> 2) << 4) + (hi << 3) + ((v & 3) << 1) + ks * 32;
                const float* kp = &Kb[(size_t)(rowb + key) * kD + h64 + kd];
                bfr.u[v] = pack2bf(kp[0], kp[1]);
            }
            s0 = __builtin_amdgcn_wmma_f32_16x16x32_bf16(false, a0.v, false, bfr.v, (short)0, s0, false, false);
            s1 = __builtin_amdgcn_wmma_f32_16x16x32_bf16(false, a1.v, false, bfr.v, (short)0, s1, false, false);
        }
        int key = knb + ln16;
        #pragma unroll
        for (int i = 0; i < 8; ++i) {
            int q0 = i + (hi << 3);
            float v0 = s0[i] * 0.125f;
            float v1 = s1[i] * 0.125f;
            if (mask) {
                if (mask[((size_t)b * kN + qbase + q0) * kN + key] == 0)      v0 = -10000.f;
                if (mask[((size_t)b * kN + qbase + 16 + q0) * kN + key] == 0) v1 = -10000.f;
            }
            Sbuf[q0 * 512 + key]        = v0;
            Sbuf[(16 + q0) * 512 + key] = v1;
        }
    }
    __syncthreads();

    // ---- phase 2: stable softmax over 512, write bf16 P ----
    {
        int row  = tid & 31;
        int part = tid >> 5;                // 4 threads per row, 128 keys each
        float* Sr = &Sbuf[row * 512 + part * 128];
        float mx = -3.0e38f;
        for (int i = 0; i < 128; ++i) mx = fmaxf(mx, Sr[i]);
        red[row * 4 + part] = mx;
        __syncthreads();
        float rm = fmaxf(fmaxf(red[row*4+0], red[row*4+1]),
                         fmaxf(red[row*4+2], red[row*4+3]));
        __syncthreads();
        float sum = 0.f;
        for (int i = 0; i < 128; ++i) { float e = __expf(Sr[i] - rm); Sr[i] = e; sum += e; }
        red[row * 4 + part] = sum;
        __syncthreads();
        float inv = 1.f / (red[row*4+0] + red[row*4+1] + red[row*4+2] + red[row*4+3]);
        unsigned* Pr = (unsigned*)&Pbuf[row * 512 + part * 128];
        for (int i = 0; i < 64; ++i)
            Pr[i] = pack2bf(Sr[2*i] * inv, Sr[2*i+1] * inv);
    }

    // ---- phase 3: O = P V  (wave -> 1 m-block x 2 dh-blocks) ----
    const int mi = w & 1;
    const int j0 = (w >> 1) << 1;
    v8f o0 = vzero8(), o1 = vzero8();
    const v4u* PQ  = (const v4u*)Pbuf;   // row stride 64 (1024B)
    const v4u* VtQ = (const v4u*)Vt;     // row stride 17 (272B)

    for (int c = 0; c < 4; ++c) {          // 128-key chunks
        __syncthreads();
        // stage V chunk transposed: Vt[dh][key], packed key-pairs
        #pragma unroll 4
        for (int i = 0; i < 32; ++i) {     // 64 dh x 64 key-pairs = 4096 pairs
            int idx = i * 128 + tid;
            int dh = idx & 63, kp = idx >> 6;
            const float* p = &Vb[(size_t)(rowb + c * 128 + 2 * kp) * kD + h64 + dh];
            ((unsigned*)Vt)[dh * 68 + kp] = pack2bf(p[0], p[kD]);
        }
        __syncthreads();
        #pragma unroll
        for (int ks = 0; ks < 4; ++ks) {
            BFrag af, b0, b1;
            int abase = (mi * 16 + ln16) * 64 + c * 16 + ks * 4 + hi;
            af.q[0] = PQ[abase];
            af.q[1] = PQ[abase + 2];
            int vb0 = (j0 * 16 + ln16) * 17 + ks * 4 + hi;
            int vb1 = ((j0 + 1) * 16 + ln16) * 17 + ks * 4 + hi;
            b0.q[0] = VtQ[vb0];  b0.q[1] = VtQ[vb0 + 2];
            b1.q[0] = VtQ[vb1];  b1.q[1] = VtQ[vb1 + 2];
            o0 = __builtin_amdgcn_wmma_f32_16x16x32_bf16(false, af.v, false, b0.v, (short)0, o0, false, false);
            o1 = __builtin_amdgcn_wmma_f32_16x16x32_bf16(false, af.v, false, b1.v, (short)0, o1, false, false);
        }
    }

    #pragma unroll
    for (int i = 0; i < 8; ++i) {
        int q = qbase + mi * 16 + i + (hi << 3);
        int col = h64 + j0 * 16 + ln16;
        AO[(size_t)(rowb + q) * kD + col]      = o0[i];
        AO[(size_t)(rowb + q) * kD + col + 16] = o1[i];
    }
}

// ============================================================
// Fused residual add + LayerNorm, one row (D=1024) per block.
// ============================================================
__global__ __launch_bounds__(256)
void ln_kernel(const float* __restrict__ hsrc, const float* __restrict__ xsrc,
               const float* __restrict__ g, const float* __restrict__ bta,
               float* __restrict__ out)
{
    __shared__ float sred[16];
    const int row = blockIdx.x;
    const int tid = threadIdx.x;
    float vals[4];
    float s = 0.f, s2 = 0.f;
    #pragma unroll
    for (int i = 0; i < 4; ++i) {
        int c = tid + i * 256;
        float v = hsrc[(size_t)row * kD + c] + xsrc[(size_t)row * kD + c];
        vals[i] = v;
        s += v; s2 += v * v;
    }
    #pragma unroll
    for (int off = 16; off; off >>= 1) {
        s  += __shfl_xor(s,  off);
        s2 += __shfl_xor(s2, off);
    }
    int wid = tid >> 5, lid = tid & 31;
    if (lid == 0) { sred[wid] = s; sred[8 + wid] = s2; }
    __syncthreads();
    if (tid == 0) {
        float ts = 0.f, ts2 = 0.f;
        for (int i = 0; i < 8; ++i) { ts += sred[i]; ts2 += sred[8 + i]; }
        sred[0] = ts; sred[8] = ts2;
    }
    __syncthreads();
    float mean = sred[0] * (1.f / kD);
    float var  = sred[8] * (1.f / kD) - mean * mean;
    float rstd = rsqrtf(var + 1e-12f);
    #pragma unroll
    for (int i = 0; i < 4; ++i) {
        int c = tid + i * 256;
        out[(size_t)row * kD + c] = g[c] * (vals[i] - mean) * rstd + bta[c];
    }
}

// ============================================================
extern "C" void kernel_launch(void* const* d_in, const int* in_sizes, int n_in,
                              void* d_out, int out_size, void* d_ws, size_t ws_size,
                              hipStream_t stream)
{
    const float* trg  = (const float*)d_in[0];
    const float* enc  = (const float*)d_in[1];
    const int*   mask = (const int*)  d_in[2];
    const float* sa_w = (const float*)d_in[3];
    const float* sa_b = (const float*)d_in[4];
    const float* ca_w = (const float*)d_in[5];
    const float* ca_b = (const float*)d_in[6];
    const float* ln_g = (const float*)d_in[7];
    const float* ln_b = (const float*)d_in[8];
    const float* w1   = (const float*)d_in[9];
    const float* b1   = (const float*)d_in[10];
    const float* w2   = (const float*)d_in[11];
    const float* b2   = (const float*)d_in[12];
    float* out = (float*)d_out;

    const size_t rowsD = (size_t)kM * kD;  // 4M floats
    float* X  = (float*)d_ws;
    float* Qb = X  + rowsD;
    float* Kb = Qb + rowsD;
    float* Vb = Kb + rowsD;
    float* T1 = Vb + rowsD;
    float* Hf = T1 + rowsD;                // kM x kFF

    hipMemcpyAsync(X, trg, sizeof(float) * rowsD, hipMemcpyDeviceToDevice, stream);

    dim3 gD(kM / 64, kD / 64);     // 64 x 16
    dim3 gF(kM / 64, kFF / 64);    // 64 x 64
    dim3 gAttn(kB * kH * (kN / kQT));  // 2048

    for (int layer = 0; layer < kL; ++layer) {
        const float* SW = sa_w + (size_t)layer * 4 * kD * kD;
        const float* SB = sa_b + (size_t)layer * 4 * kD;
        const float* CW = ca_w + (size_t)layer * 4 * kD * kD;
        const float* CB = ca_b + (size_t)layer * 4 * kD;
        const float* LG = ln_g + (size_t)layer * 3 * kD;
        const float* LB = ln_b + (size_t)layer * 3 * kD;

        // ---- self-attention ----
        gemm_bf16<<<gD, 128, 0, stream>>>(X, SW + 0 * kD * kD, SB + 0 * kD, Qb, kM, kD, kD, 0);
        gemm_bf16<<<gD, 128, 0, stream>>>(X, SW + 1 * kD * kD, SB + 1 * kD, Kb, kM, kD, kD, 0);
        gemm_bf16<<<gD, 128, 0, stream>>>(X, SW + 2 * kD * kD, SB + 2 * kD, Vb, kM, kD, kD, 0);
        attn_kernel<<<gAttn, 128, kSMEM_ATTN, stream>>>(Qb, Kb, Vb, mask, T1);
        gemm_bf16<<<gD, 128, 0, stream>>>(T1, SW + 3 * kD * kD, SB + 3 * kD, Qb, kM, kD, kD, 0);
        ln_kernel<<<kM, 256, 0, stream>>>(Qb, X, LG + 0 * kD, LB + 0 * kD, X);

        // ---- cross-attention (no mask) ----
        gemm_bf16<<<gD, 128, 0, stream>>>(X,   CW + 0 * kD * kD, CB + 0 * kD, Qb, kM, kD, kD, 0);
        gemm_bf16<<<gD, 128, 0, stream>>>(enc, CW + 1 * kD * kD, CB + 1 * kD, Kb, kM, kD, kD, 0);
        gemm_bf16<<<gD, 128, 0, stream>>>(enc, CW + 2 * kD * kD, CB + 2 * kD, Vb, kM, kD, kD, 0);
        attn_kernel<<<gAttn, 128, kSMEM_ATTN, stream>>>(Qb, Kb, Vb, nullptr, T1);
        gemm_bf16<<<gD, 128, 0, stream>>>(T1, CW + 3 * kD * kD, CB + 3 * kD, Qb, kM, kD, kD, 0);
        ln_kernel<<<kM, 256, 0, stream>>>(Qb, X, LG + 1 * kD, LB + 1 * kD, X);

        // ---- FFN ----
        gemm_bf16<<<gF, 128, 0, stream>>>(X, w1 + (size_t)layer * kD * kFF, b1 + (size_t)layer * kFF, Hf, kM, kD, kFF, 1);
        gemm_bf16<<<gD, 128, 0, stream>>>(Hf, w2 + (size_t)layer * kFF * kD, b2 + (size_t)layer * kD, T1, kM, kFF, kD, 0);
        float* dst = (layer == kL - 1) ? out : X;
        ln_kernel<<<kM, 256, 0, stream>>>(T1, X, LG + 2 * kD, LB + 2 * kD, dst);
    }
}
